// SVR_mine_19078244729275
// MI455X (gfx1250) — matrix operand order repaired and checked by reading, not compiled, not run on hardware
//
#include <hip/hip_runtime.h>

typedef __attribute__((ext_vector_type(16))) _Float16 v16h;
typedef __attribute__((ext_vector_type(8)))  _Float16 v8h;
typedef __attribute__((ext_vector_type(8)))  float    v8f;
typedef __attribute__((ext_vector_type(4)))  int      v4i;
typedef __attribute__((ext_vector_type(4)))  unsigned int v4u;

typedef __attribute__((address_space(1))) v4i as1_v4i;   // global
typedef __attribute__((address_space(3))) v4i as3_v4i;   // LDS

#define DDIM 64
#define LROW 72                       // padded LDS row (f16): 144B = 36 banks
#define NSPLIT 8
#define WAVES_PER_BLOCK 8
#define BLOCK_THREADS (WAVES_PER_BLOCK * 32)
#define CHUNK_N 128                   // n-rows staged per barrier round

#if __has_builtin(__builtin_amdgcn_global_load_async_to_lds_b128) && \
    __has_builtin(__builtin_amdgcn_s_wait_asynccnt)
#define USE_ASYNC_LDS 1
#else
#define USE_ASYNC_LDS 0
#endif

// ---------------------------------------------------------------------------
__global__ void svr_cvt_f32_to_f16(const float* __restrict__ src,
                                   _Float16* __restrict__ dst, int count) {
  int i = blockIdx.x * blockDim.x + threadIdx.x;
  if (i < count) dst[i] = (_Float16)src[i];
}

__global__ void svr_init_out(float* __restrict__ out,
                             const float* __restrict__ b, int m) {
  int i = blockIdx.x * blockDim.x + threadIdx.x;
  if (i < m) out[i] = b[0];
}

// Stage one 128x64 f16 chunk of B into LDS (padded rows).
// 256 threads x 4 x 16B segments = 16 KB.
__device__ __forceinline__
void stage_chunk(const _Float16* __restrict__ gbase, _Float16* lbase, int tid) {
#pragma unroll
  for (int j = 0; j < 4; ++j) {
    const int seg = j * BLOCK_THREADS + tid;     // 0..1023
    const int row = seg >> 3;
    const int c8  = (seg & 7) * 8;               // f16 col offset
    const _Float16* g = gbase + row * DDIM + c8;
    _Float16*       l = lbase + row * LROW + c8;
#if USE_ASYNC_LDS
    __builtin_amdgcn_global_load_async_to_lds_b128(
        (as1_v4i*)g, (as3_v4i*)l, 0, 0);
#else
    *(v4u*)l = *(const v4u*)g;                   // global ld + ds_store
#endif
  }
}

// ---------------------------------------------------------------------------
// Block = 8 waves, each owning one 16-row m-tile; all 8 share every staged
// B chunk (8x L2-traffic reuse). Double-buffered LDS, ASYNCcnt-tracked DMA.
//   WMMA(C=1.0) -> (1 + x.X);  acc += alpha[n] * (1+s)^4
// ---------------------------------------------------------------------------
__global__ void __launch_bounds__(BLOCK_THREADS)
svr_wmma_kernel(const _Float16* __restrict__ A,   // [M x 64] f16 (x_p)
                const _Float16* __restrict__ B,   // [N x 64] f16 (X)
                const float* __restrict__ alpha,  // [N]
                float* __restrict__ out,          // [M]
                int M, int N) {
  __shared__ _Float16 sB[2][CHUNK_N * LROW];      // 2 x 18 KB

  const int tid   = threadIdx.x;
  const int lane  = tid & 31;
  const int wv    = tid >> 5;
  const int nsplit = blockIdx.x % NSPLIT;
  const int mGroup = blockIdx.x / NSPLIT;
  const int m0     = (mGroup * WAVES_PER_BLOCK + wv) * 16;

  const int nPer    = N / NSPLIT;
  const int n0base  = nsplit * nPer;
  const int nChunks = nPer / CHUNK_N;             // 8

  const int hl  = lane >> 4;
  const int l16 = lane & 15;

  // A fragment (persistent): lane = row m0+l16; contiguous k order used here, same as B:
  // element p -> K = p + 16*hl  (+32 for second k-step).
  const _Float16* arow = A + (size_t)(m0 + l16) * DDIM;
  v8h a00 = *(const v8h*)(arow +      16 * hl);
  v8h a01 = *(const v8h*)(arow + 8  + 16 * hl);
  v8h a10 = *(const v8h*)(arow + 32 + 16 * hl);
  v8h a11 = *(const v8h*)(arow + 40 + 16 * hl);
  v16h a0 = __builtin_shufflevector(a00, a01, 0,1,2,3,4,5,6,7,8,9,10,11,12,13,14,15);
  v16h a1 = __builtin_shufflevector(a10, a11, 0,1,2,3,4,5,6,7,8,9,10,11,12,13,14,15);

  v8f ones;
#pragma unroll
  for (int i = 0; i < 8; ++i) ones[i] = 1.0f;
  v8f acc = {};

  // Prologue: stage chunk 0.
  stage_chunk(B + (size_t)n0base * DDIM, &sB[0][0], tid);

  for (int ck = 0; ck < nChunks; ++ck) {
    if (ck + 1 < nChunks) {
      stage_chunk(B + (size_t)(n0base + (ck + 1) * CHUNK_N) * DDIM,
                  &sB[(ck + 1) & 1][0], tid);
#if USE_ASYNC_LDS
      __builtin_amdgcn_s_wait_asynccnt(4);        // chunk ck's 4 ops done
#endif
    } else {
#if USE_ASYNC_LDS
      __builtin_amdgcn_s_wait_asynccnt(0);
#endif
    }
    __syncthreads();                              // all waves' segments visible

    const _Float16* bufB = &sB[ck & 1][0];
    const float*    aptr = alpha + n0base + ck * CHUNK_N + l16;

#pragma unroll
    for (int tt = 0; tt < CHUNK_N / 16; ++tt) {   // 8 n-tiles per chunk
      const _Float16* brow = bufB + (tt * 16 + l16) * LROW + 16 * hl;
      v16h b0 = *(const v16h*)(brow);             // K 0..31 half (ds_load)
      v16h b1 = *(const v16h*)(brow + 32);        // K 32..63 half
      const float al = aptr[tt * 16];

      v8f c = __builtin_amdgcn_wmma_f32_16x16x32_f16(false, a0, false, b0,
                                                     (short)0, ones, false, false);
      c = __builtin_amdgcn_wmma_f32_16x16x32_f16(false, a1, false, b1,
                                                 (short)0, c, false, false);
#pragma unroll
      for (int i = 0; i < 8; ++i) {
        float t2 = c[i] * c[i];
        float t4 = t2 * t2;
        acc[i] = fmaf(t4, al, acc[i]);
      }
    }
    __syncthreads();                              // buffer free for reuse
  }

  // C layout: VGPR r, lanes 0-15 -> row m0+r ; lanes 16-31 -> row m0+8+r.
#pragma unroll
  for (int r = 0; r < 8; ++r) {
    float v = acc[r];
    v += __shfl_xor(v, 1, 32);
    v += __shfl_xor(v, 2, 32);
    v += __shfl_xor(v, 4, 32);
    v += __shfl_xor(v, 8, 32);
    if (l16 == 0) atomicAdd(&out[m0 + r + 8 * hl], v);
  }
}

// ---------------------------------------------------------------------------
extern "C" void kernel_launch(void* const* d_in, const int* in_sizes, int n_in,
                              void* d_out, int out_size, void* d_ws, size_t ws_size,
                              hipStream_t stream) {
  const float* x_p   = (const float*)d_in[0];  // [M, 64]
  const float* X     = (const float*)d_in[1];  // [N, 64]
  const float* alpha = (const float*)d_in[2];  // [N]
  const float* bscal = (const float*)d_in[3];  // [1]
  float* out = (float*)d_out;                  // [M]

  const int M = in_sizes[0] / DDIM;            // 16384
  const int N = in_sizes[1] / DDIM;            // 8192

  _Float16* Ah = (_Float16*)d_ws;              // 2 MB
  _Float16* Bh = Ah + (size_t)M * DDIM;        // 1 MB

  const int cntA = M * DDIM;
  const int cntB = N * DDIM;
  svr_cvt_f32_to_f16<<<(cntA + 255) / 256, 256, 0, stream>>>(x_p, Ah, cntA);
  svr_cvt_f32_to_f16<<<(cntB + 255) / 256, 256, 0, stream>>>(X, Bh, cntB);
  svr_init_out<<<(M + 255) / 256, 256, 0, stream>>>(out, bscal, M);

  const int blocks = (M / (16 * WAVES_PER_BLOCK)) * NSPLIT;   // 1024
  svr_wmma_kernel<<<blocks, BLOCK_THREADS, 0, stream>>>(Ah, Bh, alpha, out, M, N);
}
